// SkipLSTM_69449621176507
// MI455X (gfx1250) — matrix operand order repaired and checked by reading, not compiled, not run on hardware
//
#include <hip/hip_runtime.h>
#include <hip/hip_bf16.h>

#define T_   1024
#define B_   128
#define IN_  256
#define H_   512
#define NWG  32
#define COLS_PER_WG 16          // h-columns per workgroup
#define GCOLS 64                // gate columns per workgroup (4 gates x 16)
#define KTOT (IN_ + H_)         // 768 = fused K (x | h)
#define KTILES (KTOT / 32)      // 24 k-tiles of 32
#define KTX (IN_ / 32)          // 8 k-tiles from x
#define WSTRIDE (KTOT + 8)      // 776 bf16 row stride -> 16B aligned, bank-friendly

typedef __attribute__((ext_vector_type(16))) __bf16 v16bf;
typedef __attribute__((ext_vector_type(8)))  float  v8f;
typedef __attribute__((ext_vector_type(4)))  unsigned v4u;

union ABfrag { v4u u2[2]; v16bf v; };

__device__ __forceinline__ float sigf(float x)  { return 1.0f / (1.0f + __expf(-x)); }
__device__ __forceinline__ float tanhf_(float x){ float e = __expf(2.0f * x); return 1.0f - 2.0f / (e + 1.0f); }

extern "C" __global__ void init_ws_kernel(unsigned* p, int n) {
    int i = blockIdx.x * blockDim.x + threadIdx.x;
    if (i < n) p[i] = 0u;
}

extern "C" __global__ void cvt_x_kernel(const float* __restrict__ x, __bf16* __restrict__ xb, int n) {
    int i = blockIdx.x * blockDim.x + threadIdx.x;
    if (i < n) xb[i] = (__bf16)x[i];
}

// ---- fragment loaders -------------------------------------------------------
__device__ __forceinline__ void loadB4(const __bf16* const* wp, int kt, ABfrag* b) {
#pragma unroll
    for (int g = 0; g < 4; ++g) {
        const __bf16* bp = wp[g] + 32 * kt;
        b[g].u2[0] = *(const v4u*)(bp);        // K[kb .. kb+8)
        b[g].u2[1] = *(const v4u*)(bp + 16);   // K[kb+16 .. kb+24)
    }
}

__device__ __forceinline__ void loadA1(const __bf16* ax, const __bf16* ah, int kt, ABfrag* a) {
    const __bf16* ap = (kt < KTX) ? (ax + 32 * kt) : (ah + 32 * (kt - KTX));
    a->u2[0] = *(const v4u*)(ap);              // K[kb .. kb+8)
    a->u2[1] = *(const v4u*)(ap + 16);         // K[kb+16 .. kb+24)
}

extern "C" __global__ __launch_bounds__(256)
__attribute__((amdgpu_waves_per_eu(2)))
void lstm_persist_kernel(const float* __restrict__ W_ih,
                         const float* __restrict__ W_hh,
                         const float* __restrict__ b_ih,
                         const float* __restrict__ b_hh,
                         const __bf16* __restrict__ xb,     // [T][B][IN] bf16
                         __bf16* __restrict__ hbuf,         // [2][B][H] bf16 ping-pong
                         unsigned* __restrict__ barrier,    // [0]=count [1]=generation
                         float* __restrict__ out)           // [3][B][H] f32
{
    __shared__ __align__(16) __bf16 Wk[GCOLS * WSTRIDE];    // ~97 KB of 320 KB
    __shared__ float bias_s[GCOLS];

    const int tid   = threadIdx.x;
    const int wg    = blockIdx.x;
    const int hbase = wg * COLS_PER_WG;

    // ---- prologue: convert this WG's weight slice to bf16 in LDS (once for all 1024 steps)
    for (int idx = tid; idx < GCOLS * KTOT; idx += 256) {
        int gi = idx / KTOT;
        int k  = idx - gi * KTOT;
        int gate = gi >> 4;
        int c    = gi & 15;
        int grow = gate * H_ + hbase + c;   // PyTorch gate order i,f,g,o
        float v = (k < IN_) ? W_ih[(size_t)grow * IN_ + k]
                            : W_hh[(size_t)grow * H_ + (k - IN_)];
        Wk[gi * WSTRIDE + k] = (__bf16)v;
    }
    if (tid < GCOLS) {
        int gate = tid >> 4, c = tid & 15;
        int grow = gate * H_ + hbase + c;
        bias_s[tid] = b_ih[grow] + b_hh[grow];
    }
    __syncthreads();

    // ---- per-lane WMMA fragment geometry (wave32)
    const int lane  = tid & 31;
    const int wave  = tid >> 5;            // 8 waves -> 8 M-tiles of the batch
    const int col   = lane & 15;
    const int hi    = lane >> 4;           // lane half
    const int arow  = wave * 16 + col;     // A operand batch row
    const int akoff = hi * 8;              // A K-half offset (elements)
    const int bkoff = hi * 8;              // B K-half offset (elements)
    const int crow  = wave * 16 + hi * 8;  // C/D base batch row

    float bg[4];
    const __bf16* wp[4];
#pragma unroll
    for (int g = 0; g < 4; ++g) {
        bg[g] = bias_s[g * 16 + col];
        wp[g] = &Wk[(g * 16 + col) * WSTRIDE + bkoff];
    }

    float cacc[8], hout[8];
#pragma unroll
    for (int r = 0; r < 8; ++r) { cacc[r] = 0.0f; hout[r] = 0.0f; }

    const size_t BH = (size_t)B_ * H_;

    for (int t = 0; t < T_; ++t) {
        // init accumulators with bias (broadcast down each lane's column)
        v8f acc[4];
#pragma unroll
        for (int g = 0; g < 4; ++g) {
            v8f a;
#pragma unroll
            for (int r = 0; r < 8; ++r) a[r] = bg[g];
            acc[g] = a;
        }

        const __bf16* ax = xb + ((size_t)t * B_ + arow) * IN_ + akoff;
        const __bf16* ah = hbuf + (size_t)((t + 1) & 1) * BH + (size_t)arow * H_ + akoff;

        // fused GEMM: z = [x_t | h_{t-1}] * [W_ih | W_hh]^T   (K = 768)
        // software pipeline: B (LDS) double-buffered, A (L2) triple-buffered
        ABfrag Af[3];
        ABfrag Bf[2][4];
        loadA1(ax, ah, 0, &Af[0]);
        loadB4(wp, 0, Bf[0]);
        loadA1(ax, ah, 1, &Af[1]);
#pragma unroll
        for (int kt = 0; kt < KTILES; ++kt) {
            const int cur = kt & 1;
            if (kt + 1 < KTILES) loadB4(wp, kt + 1, Bf[cur ^ 1]);
            if (kt + 2 < KTILES) loadA1(ax, ah, kt + 2, &Af[(kt + 2) % 3]);
            const ABfrag* a = &Af[kt % 3];
#pragma unroll
            for (int g = 0; g < 4; ++g)
                acc[g] = __builtin_amdgcn_wmma_f32_16x16x32_bf16(
                    false, a->v, false, Bf[cur][g].v, (short)0, acc[g], false, false);
        }

        // LSTM cell update: each lane holds all four gates for its h-column
        __bf16* hw = hbuf + (size_t)(t & 1) * BH;
#pragma unroll
        for (int r = 0; r < 8; ++r) {
            float iv = sigf(acc[0][r]);
            float fv = sigf(acc[1][r]);
            float gv = tanhf_(acc[2][r]);
            float ov = sigf(acc[3][r]);
            float cv = fv * cacc[r] + iv * gv;
            cacc[r] = cv;
            float hv = ov * tanhf_(cv);
            hout[r] = hv;
            hw[(size_t)(crow + r) * H_ + hbase + col] = (__bf16)hv;
        }

        // ---- device-scope barrier (persistent grid, sense via generation counter)
        __syncthreads();
        if (tid == 0) {
            __threadfence();                                  // publish h stores
            volatile unsigned* vgen = barrier + 1;
            unsigned g0 = *vgen;
            if (atomicAdd(barrier, 1u) == NWG - 1) {
                atomicExch(barrier, 0u);
                __threadfence();
                atomicAdd(barrier + 1, 1u);
            } else {
                while (*vgen == g0) { __builtin_amdgcn_s_sleep(1); }
            }
            __threadfence();
        }
        __syncthreads();
    }

    // ---- epilogue: (output, h_n, c_n) all from the last step, fp32
#pragma unroll
    for (int r = 0; r < 8; ++r) {
        size_t row = (size_t)(crow + r);
        size_t c0  = (size_t)hbase + col;
        out[row * H_ + c0]           = hout[r];
        out[BH + row * H_ + c0]      = hout[r];
        out[2 * BH + row * H_ + c0]  = cacc[r];
    }
}

extern "C" void kernel_launch(void* const* d_in, const int* in_sizes, int n_in,
                              void* d_out, int out_size, void* d_ws, size_t ws_size,
                              hipStream_t stream) {
    (void)in_sizes; (void)n_in; (void)out_size; (void)ws_size;

    const float* x    = (const float*)d_in[0];
    const float* W_ih = (const float*)d_in[1];
    const float* W_hh = (const float*)d_in[2];
    const float* b_ih = (const float*)d_in[3];
    const float* b_hh = (const float*)d_in[4];
    float* out = (float*)d_out;

    char* ws = (char*)d_ws;
    // ws layout: [0,256): barrier; [256, 256+256KB): h ping-pong (bf16);
    //            then x in bf16 (64 MB, fits in 192 MB L2)
    unsigned* bar  = (unsigned*)ws;
    __bf16*   hbuf = (__bf16*)(ws + 256);
    const size_t hbytes = 2 * (size_t)B_ * H_ * sizeof(__bf16);   // 256 KB
    __bf16*   xb   = (__bf16*)(ws + 256 + hbytes);

    // zero barrier state + both h buffers (h0 = 0); runs every call -> deterministic
    int zn = (int)((256 + hbytes) / 4);
    init_ws_kernel<<<(zn + 255) / 256, 256, 0, stream>>>((unsigned*)ws, zn);

    int xn = T_ * B_ * IN_;
    cvt_x_kernel<<<(xn + 255) / 256, 256, 0, stream>>>(x, xb, xn);

    lstm_persist_kernel<<<NWG, 256, 0, stream>>>(W_ih, W_hh, b_ih, b_hh, xb, hbuf, bar, out);
}
